// GatedDeltaNetLayer_75892072120388
// MI455X (gfx1250) — matrix operand order repaired and checked by reading, not compiled, not run on hardware
//
#include <hip/hip_runtime.h>
#include <hip/hip_bf16.h>

typedef unsigned short u16;
typedef __attribute__((ext_vector_type(16))) __bf16 v16bf;
typedef __attribute__((ext_vector_type(8)))  float  v8f;

#define B_   4
#define T_   8192
#define D_   768
#define H_   12
#define ROWS_ (B_*T_)   /* 32768 */

struct alignas(16) U4 { unsigned x, y, z, w; };
union Frag { v16bf v; U4 q[2]; };

__device__ __forceinline__ u16 f2bf(float f) {
  unsigned u = __float_as_uint(f);
  u += 0x7fffu + ((u >> 16) & 1u);
  return (u16)(u >> 16);
}

__device__ __forceinline__ v8f wmma_bf16(v16bf a, v16bf b, v8f c) {
  return __builtin_amdgcn_wmma_f32_16x16x32_bf16(false, a, false, b, (short)0, c,
                                                 false, false);
}

// Load a 16-element bf16 fragment as two 16B chunks, second chunk `step` u16s later.
// A-fragment: step=16 (K runs koff..koff+7 and koff+16..koff+23)
// B-fragment: step=16 (K runs ks0..ks0+7 and ks0+16..ks0+23)
__device__ __forceinline__ v16bf frag_ld2(const u16* p, int step) {
  Frag f;
  f.q[0] = *(const U4*)(p);
  f.q[1] = *(const U4*)(p + step);
  return f.v;
}

__device__ __forceinline__ v8f zero8() {
  v8f z = {0.f,0.f,0.f,0.f,0.f,0.f,0.f,0.f};
  return z;
}

// ---------------------------------------------------------------------------
// Kernel 1: convert + transpose weights to bf16.
// wkT/wvT: [out=768][in=768], woT: [out=D][in=HV], wbgT: [32][768] (12 Wb rows,
// 12 Wg rows, 8 zero pad rows)
// ---------------------------------------------------------------------------
__global__ __launch_bounds__(256)
void convert_weights_kernel(const float* __restrict__ Wk, const float* __restrict__ Wv,
                            const float* __restrict__ Wo, const float* __restrict__ Wb,
                            const float* __restrict__ Wg,
                            u16* __restrict__ wkT, u16* __restrict__ wvT,
                            u16* __restrict__ woT, u16* __restrict__ wbgT) {
  const int N = D_ * D_;
  int idx = blockIdx.x * 256 + threadIdx.x;
  if (idx < N) {
    int o = idx / D_, i = idx - o * D_;
    wkT[idx] = f2bf(Wk[i * D_ + o]);
  } else if (idx < 2 * N) {
    int j = idx - N;
    int o = j / D_, i = j - o * D_;
    wvT[j] = f2bf(Wv[i * D_ + o]);
  } else if (idx < 3 * N) {
    int j = idx - 2 * N;
    int d = j / D_, hv = j - d * D_;
    woT[j] = f2bf(Wo[hv * D_ + d]);
  } else if (idx < 3 * N + 32 * D_) {
    int j = idx - 3 * N;
    int r = j / D_, i = j - r * D_;
    float val = 0.f;
    if (r < 12)       val = Wb[i * H_ + r];
    else if (r < 24)  val = Wg[i * H_ + (r - 12)];
    wbgT[j] = f2bf(val);
  }
}

// ---------------------------------------------------------------------------
// Kernel 2: fused RMSNorm + projections (k raw, v, beta, g).
// One workgroup = 64 rows.  xn staged once in LDS (bf16), then WMMA GEMM.
// Each wave computes a 64x32 output slab (2 column tiles) so every A-fragment
// LDS read feeds two WMMAs; K-loop unroll capped to avoid register spills.
// ---------------------------------------------------------------------------
__global__ __launch_bounds__(256)
void rms_proj_kernel(const float* __restrict__ x, const float* __restrict__ wn,
                     const u16* __restrict__ wkT, const u16* __restrict__ wvT,
                     const u16* __restrict__ wbgT,
                     const float* __restrict__ bb, const float* __restrict__ bg,
                     float* __restrict__ kraw, float* __restrict__ vraw,
                     float* __restrict__ betab, float* __restrict__ gb) {
  __shared__ u16   sXn[64 * 776];
  __shared__ float sScale[64];
  __shared__ float sRed[64][4];

  const int tid = threadIdx.x;
  const int rowbase = blockIdx.x * 64;
  const int r = tid >> 2, q = tid & 3;
  const float* xr = x + (size_t)(rowbase + r) * D_;

  // Phase 1: per-row sum of squares (4 threads per row)
  float ss = 0.f;
  for (int j = 0; j < 48; ++j) {
    float4 v = *(const float4*)(xr + q * 192 + j * 4);
    ss += v.x * v.x + v.y * v.y + v.z * v.z + v.w * v.w;
  }
  sRed[r][q] = ss;
  __syncthreads();
  if (q == 0)
    sScale[r] = rsqrtf((sRed[r][0] + sRed[r][1] + sRed[r][2] + sRed[r][3]) *
                           (1.f / (float)D_) + 1e-6f);
  __syncthreads();

  // Phase 2: xn = x * rms_scale * w_norm -> bf16 LDS tile [64][776]
  const float sc = sScale[r];
  for (int j = 0; j < 48; ++j) {
    int c = q * 192 + j * 4;
    float4 v = *(const float4*)(xr + c);
    float4 w = *(const float4*)(wn + c);
    u16* d = sXn + r * 776 + c;
    d[0] = f2bf(v.x * sc * w.x); d[1] = f2bf(v.y * sc * w.y);
    d[2] = f2bf(v.z * sc * w.z); d[3] = f2bf(v.w * sc * w.w);
  }
  __syncthreads();

  const int wave = tid >> 5, lane = tid & 31;
  const int n16 = lane & 15, hi = lane >> 4;
  const int koffA = hi * 8, ks0B = hi * 8, mrow0 = hi * 8;

  // Phase 3: k and v projections. 6 slabs of 256 columns (0..2 -> k, 3..5 -> v);
  // each wave owns 32 columns (two 16-col tiles).
  for (int ni = 0; ni < 6; ++ni) {
    const u16* WT = (ni < 3) ? wkT : wvT;
    float* dst    = (ni < 3) ? kraw : vraw;
    const int ncol0 = (ni % 3) * 256 + wave * 32 + n16;
    const int ncol1 = ncol0 + 16;
    v8f acc0[4], acc1[4];
    #pragma unroll
    for (int mt = 0; mt < 4; ++mt) { acc0[mt] = zero8(); acc1[mt] = zero8(); }
    const u16* wb0 = WT + (size_t)ncol0 * D_ + ks0B;
    const u16* wb1 = WT + (size_t)ncol1 * D_ + ks0B;
    #pragma unroll 2
    for (int kk = 0; kk < 24; ++kk) {
      __builtin_prefetch(wb0 + (kk + 4) * 32, 0, 1);
      v16bf fb0 = frag_ld2(wb0 + kk * 32, 16);
      v16bf fb1 = frag_ld2(wb1 + kk * 32, 16);
      #pragma unroll
      for (int mt = 0; mt < 4; ++mt) {
        v16bf fa = frag_ld2(sXn + (n16 + mt * 16) * 776 + kk * 32 + koffA, 16);
        acc0[mt] = wmma_bf16(fa, fb0, acc0[mt]);
        acc1[mt] = wmma_bf16(fa, fb1, acc1[mt]);
      }
    }
    #pragma unroll
    for (int mt = 0; mt < 4; ++mt)
      #pragma unroll
      for (int e = 0; e < 8; ++e) {
        size_t row = (size_t)(rowbase + mt * 16 + mrow0 + e);
        dst[row * D_ + ncol0] = acc0[mt][e];
        dst[row * D_ + ncol1] = acc1[mt][e];
      }
  }

  // Phase 4: beta / g (padded 32-column GEMM on waves 0..1, sigmoid on store)
  if (wave < 2) {
    const int ncol = wave * 16 + n16;  // 0..31
    v8f acc[4];
    #pragma unroll
    for (int mt = 0; mt < 4; ++mt) acc[mt] = zero8();
    const u16* wb_ = wbgT + (size_t)ncol * D_ + ks0B;
    #pragma unroll 2
    for (int kk = 0; kk < 24; ++kk) {
      v16bf fb = frag_ld2(wb_ + kk * 32, 16);
      #pragma unroll
      for (int mt = 0; mt < 4; ++mt) {
        v16bf fa = frag_ld2(sXn + (n16 + mt * 16) * 776 + kk * 32 + koffA, 16);
        acc[mt] = wmma_bf16(fa, fb, acc[mt]);
      }
    }
    if (ncol < 24) {
      const bool isb = ncol < 12;
      const int hh = isb ? ncol : ncol - 12;
      const float bias = isb ? bb[hh] : bg[hh];
      float* dst = isb ? betab : gb;
      #pragma unroll
      for (int mt = 0; mt < 4; ++mt)
        #pragma unroll
        for (int e = 0; e < 8; ++e) {
          float t = acc[mt][e] + bias;
          dst[(size_t)(rowbase + mt * 16 + mrow0 + e) * H_ + hh] =
              1.f / (1.f + __expf(-t));
        }
    }
  }
}

// ---------------------------------------------------------------------------
// Kernel 3: L2-normalize k per (b,t,h) 64-vector, emit bf16 khat.
// ---------------------------------------------------------------------------
__global__ __launch_bounds__(256)
void knorm_kernel(const float* __restrict__ kraw, u16* __restrict__ khat) {
  const int tid = threadIdx.x;
  const int row = blockIdx.x * 8 + (tid >> 5);
  const int lane = tid & 31;
  const float* kp = kraw + (size_t)row * 64;
  float a = kp[lane], b = kp[lane + 32];
  float s = a * a + b * b;
  #pragma unroll
  for (int off = 16; off; off >>= 1) s += __shfl_xor(s, off, 32);
  float scl = 1.f / fmaxf(sqrtf(s), 1e-12f);
  khat[(size_t)row * 64 + lane]      = f2bf(a * scl);
  khat[(size_t)row * 64 + lane + 32] = f2bf(b * scl);
}

// ---------------------------------------------------------------------------
// Kernel 4: chunked gated delta-rule scan.  One workgroup per (b,h).
// Chunk C=64:  Gram = Khat Khat^T,  Q = Khat S0,
//   u_t = beta_t (v_t - gprev_t q_t) - sum_{i<t} A'[t,i] u_i   (triangular solve)
//   Out = gamma .* Q + M U,          S^T = gamma_C S^T + (c.*U)^T Khat
// ---------------------------------------------------------------------------
__global__ __launch_bounds__(256)
void scan_kernel(const u16* __restrict__ khat, const float* __restrict__ vraw,
                 const float* __restrict__ betab, const float* __restrict__ gin,
                 float* __restrict__ outs, float* __restrict__ fstate) {
  __shared__ u16 sK  [64 * 72];   // Khat [t][k]
  __shared__ u16 sKT [64 * 72];   // Khat^T [k][t]
  __shared__ u16 sST [64 * 72];   // S^T bf16 [v][k]
  __shared__ u16 sUT [64 * 72];   // U^T bf16 [v][i]
  __shared__ u16 sUcT[64 * 72];   // (c_i * U)^T bf16 [v][i]
  __shared__ u16 sM  [64 * 72];   // M bf16 [t][i]
  __shared__ u16 sAlow[32 * 40];  // A'[32:64, 0:32] bf16
  __shared__ float sGQ [64 * 65]; // Gram, then Q (Gram fully consumed first)
  __shared__ float sA  [64 * 65]; // A' f32
  __shared__ float sU  [64 * 65]; // rhs -> U f32
  __shared__ float sSTf[64 * 65]; // S^T f32 master
  __shared__ float sG[64], sBeta[64], sGam[64];

  const int tid = threadIdx.x;
  const int bh = blockIdx.x, b = bh / H_, h = bh % H_;
  const int wave = tid >> 5, lane = tid & 31;
  const int n16 = lane & 15, hi = lane >> 4;
  const int koffA = hi * 8, ks0B = hi * 8, mrow0 = hi * 8;

  for (int i = tid; i < 64 * 65; i += 256) sSTf[i] = 0.f;
  for (int i = tid; i < 64 * 72; i += 256) sST[i] = 0;
  __syncthreads();

  for (int t0 = 0; t0 < T_; t0 += 64) {
    // -- load khat chunk (+ transpose), g, beta ------------------------------
    for (int idx = tid; idx < 2048; idx += 256) {
      int t = idx >> 5, c2 = (idx & 31) * 2;
      unsigned u = *(const unsigned*)(khat +
          ((size_t)((b * T_ + t0 + t) * H_ + h)) * 64 + c2);
      *(unsigned*)(sK + t * 72 + c2) = u;
      sKT[c2 * 72 + t]       = (u16)u;
      sKT[(c2 + 1) * 72 + t] = (u16)(u >> 16);
    }
    if (tid < 64) {
      int gi = (b * T_ + t0 + tid) * H_ + h;
      sG[tid] = gin[gi];
      sBeta[tid] = betab[gi];
    }
    __syncthreads();
    if (tid == 0) {
      float p = 1.f;
      for (int t = 0; t < 64; ++t) { p *= sG[t]; sGam[t] = p; }
    }
    __syncthreads();

    // -- Gram = Khat Khat^T --------------------------------------------------
    for (int w2 = 0; w2 < 2; ++w2) {
      int tile = wave * 2 + w2, mt = tile >> 2, nt = tile & 3;
      v8f acc = zero8();
      #pragma unroll
      for (int kk = 0; kk < 2; ++kk) {
        v16bf fa = frag_ld2(sK + (n16 + mt * 16) * 72 + kk * 32 + koffA, 16);
        v16bf fb = frag_ld2(sK + (nt * 16 + n16) * 72 + kk * 32 + ks0B, 16);
        acc = wmma_bf16(fa, fb, acc);
      }
      #pragma unroll
      for (int e = 0; e < 8; ++e)
        sGQ[(mt * 16 + mrow0 + e) * 65 + nt * 16 + n16] = acc[e];
    }
    __syncthreads();

    // -- A' (strict lower, f32 + bf16 low block), M (lower incl diag, bf16) --
    for (int idx = tid; idx < 4096; idx += 256) {
      int t = idx >> 6, i = idx & 63;
      float gr = sGQ[t * 65 + i];
      float gp = t ? sGam[t - 1] : 1.f;
      float ap = (i < t) ? sBeta[t] * (gp / sGam[i]) * gr : 0.f;
      sA[t * 65 + i] = ap;
      sM[t * 72 + i] = (i <= t) ? f2bf((sGam[t] / sGam[i]) * gr) : (u16)0;
      if (t >= 32 && i < 32) sAlow[(t - 32) * 40 + i] = f2bf(ap);
    }
    __syncthreads();

    // -- Q = Khat @ S0 (overwrites Gram) -------------------------------------
    for (int w2 = 0; w2 < 2; ++w2) {
      int tile = wave * 2 + w2, mt = tile >> 2, nt = tile & 3;
      v8f acc = zero8();
      #pragma unroll
      for (int kk = 0; kk < 2; ++kk) {
        v16bf fa = frag_ld2(sK + (n16 + mt * 16) * 72 + kk * 32 + koffA, 16);
        v16bf fb = frag_ld2(sST + (nt * 16 + n16) * 72 + kk * 32 + ks0B, 16);
        acc = wmma_bf16(fa, fb, acc);
      }
      #pragma unroll
      for (int e = 0; e < 8; ++e)
        sGQ[(mt * 16 + mrow0 + e) * 65 + nt * 16 + n16] = acc[e];
    }
    __syncthreads();

    // -- rhs: U[t][v] = beta_t * (v - gprev_t * Q) ---------------------------
    for (int idx = tid; idx < 1024; idx += 256) {
      int t = idx >> 4, c4 = (idx & 15) * 4;
      float4 vv = *(const float4*)(vraw +
          ((size_t)((b * T_ + t0 + t) * H_ + h)) * 64 + c4);
      float bt = sBeta[t], gp = t ? sGam[t - 1] : 1.f;
      sU[t * 65 + c4 + 0] = bt * (vv.x - gp * sGQ[t * 65 + c4 + 0]);
      sU[t * 65 + c4 + 1] = bt * (vv.y - gp * sGQ[t * 65 + c4 + 1]);
      sU[t * 65 + c4 + 2] = bt * (vv.z - gp * sGQ[t * 65 + c4 + 2]);
      sU[t * 65 + c4 + 3] = bt * (vv.w - gp * sGQ[t * 65 + c4 + 3]);
    }
    __syncthreads();

    // -- triangular solve, diagonal block 0 (columns independent) ------------
    if (tid < 64) {
      for (int t = 1; t < 32; ++t) {
        float acc = sU[t * 65 + tid];
        for (int i = 0; i < t; ++i)
          acc = fmaf(-sA[t * 65 + i], sU[i * 65 + tid], acc);
        sU[t * 65 + tid] = acc;
      }
    }
    __syncthreads();
    for (int idx = tid; idx < 2048; idx += 256) {      // U^T lower half
      int v = idx >> 5, i = idx & 31;
      sUT[v * 72 + i] = f2bf(sU[i * 65 + v]);
    }
    __syncthreads();
    {   // off-diagonal: U[32:64] -= A'[32:64,0:32] @ U[0:32]   (WMMA, K=32)
      int mt = wave >> 2, nt = wave & 3;
      v16bf fa = frag_ld2(sAlow + (n16 + mt * 16) * 40 + koffA, 16);
      v16bf fb = frag_ld2(sUT + (nt * 16 + n16) * 72 + ks0B, 16);
      v8f acc = wmma_bf16(fa, fb, zero8());
      #pragma unroll
      for (int e = 0; e < 8; ++e)
        sU[(32 + mt * 16 + mrow0 + e) * 65 + nt * 16 + n16] -= acc[e];
    }
    __syncthreads();
    if (tid < 64) {   // diagonal block 1
      for (int t = 33; t < 64; ++t) {
        float acc = sU[t * 65 + tid];
        for (int i = 32; i < t; ++i)
          acc = fmaf(-sA[t * 65 + i], sU[i * 65 + tid], acc);
        sU[t * 65 + tid] = acc;
      }
    }
    __syncthreads();
    for (int idx = tid; idx < 2048; idx += 256) {      // U^T upper half
      int v = idx >> 5, i = 32 + (idx & 31);
      sUT[v * 72 + i] = f2bf(sU[i * 65 + v]);
    }
    const float gC = sGam[63];
    for (int idx = tid; idx < 4096; idx += 256) {      // (c_i * U)^T
      int v = idx >> 6, i = idx & 63;
      sUcT[v * 72 + i] = f2bf((gC / sGam[i]) * sU[i * 65 + v]);
    }
    __syncthreads();

    // -- Out = gamma .* Q + M @ U  -> global outs ----------------------------
    float* outp = outs + ((size_t)((b * T_ + t0) * H_ + h)) * 64;
    for (int w2 = 0; w2 < 2; ++w2) {
      int tile = wave * 2 + w2, mt = tile >> 2, nt = tile & 3;
      v8f acc = zero8();
      #pragma unroll
      for (int kk = 0; kk < 2; ++kk) {
        v16bf fa = frag_ld2(sM + (n16 + mt * 16) * 72 + kk * 32 + koffA, 16);
        v16bf fb = frag_ld2(sUT + (nt * 16 + n16) * 72 + kk * 32 + ks0B, 16);
        acc = wmma_bf16(fa, fb, acc);
      }
      int vcol = nt * 16 + n16;
      #pragma unroll
      for (int e = 0; e < 8; ++e) {
        int trow = mt * 16 + mrow0 + e;
        outp[(size_t)trow * (H_ * 64) + vcol] =
            acc[e] + sGam[trow] * sGQ[trow * 65 + vcol];
      }
    }

    // -- state: S^T = gC * S^T + UcT @ KhatT ---------------------------------
    for (int w2 = 0; w2 < 2; ++w2) {
      int tile = wave * 2 + w2, mt = tile >> 2, nt = tile & 3;
      int kcol = nt * 16 + n16;
      v8f acc;
      #pragma unroll
      for (int e = 0; e < 8; ++e)
        acc[e] = gC * sSTf[(mt * 16 + mrow0 + e) * 65 + kcol];
      #pragma unroll
      for (int kk = 0; kk < 2; ++kk) {
        v16bf fa = frag_ld2(sUcT + (n16 + mt * 16) * 72 + kk * 32 + koffA, 16);
        v16bf fb = frag_ld2(sKT + kcol * 72 + kk * 32 + ks0B, 16);
        acc = wmma_bf16(fa, fb, acc);
      }
      #pragma unroll
      for (int e = 0; e < 8; ++e) {
        int vrow = mt * 16 + mrow0 + e;
        sSTf[vrow * 65 + kcol] = acc[e];
        sST[vrow * 72 + kcol] = f2bf(acc[e]);
      }
    }
    __syncthreads();
  }

  // final_state [B,H,K,V] = S (un-transpose)
  float* fp = fstate + (size_t)(b * H_ + h) * 64 * 64;
  for (int idx = tid; idx < 4096; idx += 256) {
    int k = idx >> 6, v = idx & 63;
    fp[k * 64 + v] = sSTf[v * 65 + k];
  }
}

// ---------------------------------------------------------------------------
// Kernel 5: out = x + outs @ Wo  (WMMA, residual fused). Same 64x32-per-wave
// tiling and unroll cap as the input projection.
// ---------------------------------------------------------------------------
__global__ __launch_bounds__(256)
void out_proj_kernel(const float* __restrict__ outsm, const u16* __restrict__ woT,
                     const float* __restrict__ x, float* __restrict__ dout) {
  __shared__ u16 sY[64 * 776];
  const int tid = threadIdx.x;
  const int rowbase = blockIdx.x * 64;
  const int r = tid >> 2, q = tid & 3;
  const float* yr = outsm + (size_t)(rowbase + r) * D_;
  for (int j = 0; j < 48; ++j) {
    int c = q * 192 + j * 4;
    float4 v = *(const float4*)(yr + c);
    u16* d = sY + r * 776 + c;
    d[0] = f2bf(v.x); d[1] = f2bf(v.y); d[2] = f2bf(v.z); d[3] = f2bf(v.w);
  }
  __syncthreads();

  const int wave = tid >> 5, lane = tid & 31;
  const int n16 = lane & 15, hi = lane >> 4;
  const int koffA = hi * 8, ks0B = hi * 8, mrow0 = hi * 8;

  for (int ni = 0; ni < 3; ++ni) {
    const int ncol0 = ni * 256 + wave * 32 + n16;
    const int ncol1 = ncol0 + 16;
    v8f acc0[4], acc1[4];
    #pragma unroll
    for (int mt = 0; mt < 4; ++mt) { acc0[mt] = zero8(); acc1[mt] = zero8(); }
    const u16* wb0 = woT + (size_t)ncol0 * D_ + ks0B;
    const u16* wb1 = woT + (size_t)ncol1 * D_ + ks0B;
    #pragma unroll 2
    for (int kk = 0; kk < 24; ++kk) {
      __builtin_prefetch(wb0 + (kk + 4) * 32, 0, 1);
      v16bf fb0 = frag_ld2(wb0 + kk * 32, 16);
      v16bf fb1 = frag_ld2(wb1 + kk * 32, 16);
      #pragma unroll
      for (int mt = 0; mt < 4; ++mt) {
        v16bf fa = frag_ld2(sY + (n16 + mt * 16) * 776 + kk * 32 + koffA, 16);
        acc0[mt] = wmma_bf16(fa, fb0, acc0[mt]);
        acc1[mt] = wmma_bf16(fa, fb1, acc1[mt]);
      }
    }
    #pragma unroll
    for (int mt = 0; mt < 4; ++mt)
      #pragma unroll
      for (int e = 0; e < 8; ++e) {
        size_t row = (size_t)(rowbase + mt * 16 + mrow0 + e);
        dout[row * D_ + ncol0] = x[row * D_ + ncol0] + acc0[mt][e];
        dout[row * D_ + ncol1] = x[row * D_ + ncol1] + acc1[mt][e];
      }
  }
}

// ---------------------------------------------------------------------------
extern "C" void kernel_launch(void* const* d_in, const int* in_sizes, int n_in,
                              void* d_out, int out_size, void* d_ws, size_t ws_size,
                              hipStream_t stream) {
  const float* x  = (const float*)d_in[0];
  const float* wn = (const float*)d_in[1];
  const float* Wk = (const float*)d_in[2];
  const float* Wv = (const float*)d_in[3];
  const float* Wo = (const float*)d_in[4];
  const float* Wb = (const float*)d_in[5];
  const float* bb = (const float*)d_in[6];
  const float* Wg = (const float*)d_in[7];
  const float* bg = (const float*)d_in[8];
  (void)in_sizes; (void)n_in; (void)out_size; (void)ws_size;

  char* ws = (char*)d_ws;
  const size_t NW = (size_t)D_ * D_;
  u16* wkT  = (u16*)ws;
  u16* wvT  = wkT + NW;
  u16* woT  = wvT + NW;
  u16* wbgT = woT + NW;
  size_t off = (((size_t)((char*)(wbgT + 32 * D_) - ws)) + 255) & ~(size_t)255;
  float* kraw = (float*)(ws + off); off += (size_t)ROWS_ * D_ * 4;  // later: outs
  float* vraw = (float*)(ws + off); off += (size_t)ROWS_ * D_ * 4;
  u16*   khat = (u16*)  (ws + off); off += (size_t)ROWS_ * D_ * 2;
  float* betab= (float*)(ws + off); off += (size_t)ROWS_ * H_ * 4;
  float* gb   = (float*)(ws + off); off += (size_t)ROWS_ * H_ * 4;

  float* outF = (float*)d_out;
  float* fst  = outF + (size_t)ROWS_ * D_;

  const int convN = (int)(3 * NW + 32 * D_);
  convert_weights_kernel<<<(convN + 255) / 256, 256, 0, stream>>>(
      Wk, Wv, Wo, Wb, Wg, wkT, wvT, woT, wbgT);
  rms_proj_kernel<<<ROWS_ / 64, 256, 0, stream>>>(
      x, wn, wkT, wvT, wbgT, bb, bg, kraw, vraw, betab, gb);
  knorm_kernel<<<(ROWS_ * H_) / 8, 256, 0, stream>>>(kraw, khat);
  scan_kernel<<<B_ * H_, 256, 0, stream>>>(khat, vraw, betab, gb,
                                           /*outs=*/kraw, fst);
  out_proj_kernel<<<ROWS_ / 64, 256, 0, stream>>>(kraw, woT, x, outF);
}
